// ConvDownsample2D_30803505447385
// MI455X (gfx1250) — matrix operand order repaired and checked by reading, not compiled, not run on hardware
//
#include <hip/hip_runtime.h>

// ---------------------------------------------------------------------------
// ConvDownsample2D for MI455X (gfx1250, wave32, WMMA)
//   stage 1: separable 4x4 blur ([1,3,3,1]^2 / 64), pad 2, stride 1  (VALU, f32)
//   stage 2: 3x3x32->64 conv, stride 2, VALID                         (WMMA bf16, f32 acc)
// Memory-bound design: x read once (537MB) + out written once (268MB) ~= 35us
// floor @ 23.3TB/s; bf16 16x16x32 WMMA keeps the 43 GFLOP of math under that.
// ---------------------------------------------------------------------------

typedef __attribute__((ext_vector_type(16))) __bf16 v16bf;
typedef __attribute__((ext_vector_type(8)))  __bf16 v8bf;
typedef __attribute__((ext_vector_type(8)))  float  v8f;

#define H     512
#define W     512
#define CIN   32
#define COUT  64
#define OH    256
#define OW    256
#define BATCH 16

// per-workgroup output tile
#define TOH 4
#define TOW 16

// blurred-activation (y) tile in LDS: 9 rows x 33 cols x 32 ch (bf16)
// slot stride 40 bf16 = 80 B  -> every WMMA A-fragment piece is 16B aligned
#define YROWS 9
#define YCOLS 33
#define YSLOT 40
// vertical-pass intermediate (f32): 9 rows x 36 x-cols x 32 ch
#define VROWS 9
#define VCOLS 36

union Frag { v16bf v; v8bf h[2]; };

// Pack conv_w (3,3,32,64) f32 -> bf16 in WMMA B-matrix layout.
// Layout: wpk[((s*4 + nt)*32 + lane)*16 + j]
//   s    = kh*3+kw (K-chunk of 32 channels), nt = Cout/16 tile
//   lane: n = lane&15 (cout within tile), kb = lane>>4, ci = kb*16 + j
__global__ __launch_bounds__(256) void pack_weights_kernel(
    const float* __restrict__ conv_w, __bf16* __restrict__ wpk) {
  int idx = blockIdx.x * blockDim.x + threadIdx.x;
  if (idx >= 9 * 4 * 32 * 16) return;
  int j  = idx & 15;
  int l  = (idx >> 4) & 31;
  int nt = (idx >> 9) & 3;
  int s  = idx >> 11;
  int kb = l >> 4;
  int n  = l & 15;
  int ci   = kb * 16 + j;
  int cout = nt * 16 + n;
  wpk[idx] = (__bf16)(conv_w[((size_t)s * CIN + ci) * COUT + cout]);
}

__global__ __launch_bounds__(128) void conv_downsample_kernel(
    const float* __restrict__ x, const __bf16* __restrict__ wpk,
    const float* __restrict__ conv_b, float* __restrict__ out) {
  __shared__ float  v_lds[VROWS * VCOLS * CIN];   // 41,472 B
  __shared__ __bf16 y_lds[YROWS * YCOLS * YSLOT]; // 23,760 B  (total 65,232 B)

  const int tid = threadIdx.x;
  const int ow0 = blockIdx.x * TOW;
  const int oh0 = blockIdx.y * TOH;
  const int b   = blockIdx.z;

  const int xr0 = 2 * oh0 - 2;  // first x row needed (pad_t = 2)
  const int xc0 = 2 * ow0 - 2;  // first x col needed (pad_l = 2)

  // ---- Phase 1: vertical 4-tap blur, global x -> v_lds (f32) ----
  // 36 x-cols * 32 ch = 1152 columns; 128 threads -> 9 columns each.
  for (int col = tid; col < VCOLS * CIN; col += 128) {
    int jx = col >> 5;
    int c  = col & 31;
    int xc = xc0 + jx;
    bool cok = (xc >= 0) & (xc < W);
    float xw[12];
#pragma unroll
    for (int r = 0; r < 12; ++r) {
      int xr = xr0 + r;
      float val = 0.0f;
      if (cok && xr >= 0 && xr < H)
        val = x[(((size_t)b * H + xr) * W + xc) * CIN + c];
      xw[r] = val;
    }
#pragma unroll
    for (int i = 0; i < VROWS; ++i) {
      float v = xw[i] + 3.0f * xw[i + 1] + 3.0f * xw[i + 2] + xw[i + 3];
      v_lds[(i * VCOLS + jx) * CIN + c] = v;
    }
  }
  __syncthreads();

  // ---- Phase 2: horizontal 4-tap blur, v_lds -> y_lds (bf16), scale 1/64 ----
  for (int i = 0; i < YROWS; ++i) {
    for (int e = tid; e < YCOLS * CIN; e += 128) {
      int jy = e >> 5;
      int c  = e & 31;
      const float* vp = &v_lds[(i * VCOLS + jy) * CIN + c];
      float y = (vp[0] + 3.0f * vp[CIN] + 3.0f * vp[2 * CIN] + vp[3 * CIN])
                * (1.0f / 64.0f);
      y_lds[(i * YCOLS + jy) * YSLOT + c] = (__bf16)y;
    }
  }
  __syncthreads();

  // ---- Phase 3: implicit GEMM, M=16 pixels/wave, K=288, N=64 ----
  const int wave = tid >> 5;   // output row within tile (0..3)
  const int lane = tid & 31;
  const int m    = lane & 15;  // A row / D col
  const int kb   = lane >> 4;  // K-half select

  v8f acc0 = {}, acc1 = {}, acc2 = {}, acc3 = {};

#pragma unroll
  for (int s = 0; s < 9; ++s) {
    const int kh2 = s / 3, kw2 = s % 3;
    // A fragment: 16 output pixels (m -> ow), K = 32 channels of tap (kh2,kw2).
    // k order used here (same as wpk): lane-half kb holds K in [16kb,16kb+16).
    const int slot = (2 * wave + kh2) * YCOLS + (2 * m + kw2);
    const __bf16* ap = &y_lds[slot * YSLOT];
    Frag a;
    a.h[0] = *(const v8bf*)(ap + 16 * kb);
    a.h[1] = *(const v8bf*)(ap + 16 * kb + 8);

    const __bf16* bp = wpk + ((size_t)(s * 4) * 32 + lane) * 16;
    Frag b0, b1, b2, b3;
    b0.h[0] = *(const v8bf*)(bp + 0 * 512);
    b0.h[1] = *(const v8bf*)(bp + 0 * 512 + 8);
    b1.h[0] = *(const v8bf*)(bp + 1 * 512);
    b1.h[1] = *(const v8bf*)(bp + 1 * 512 + 8);
    b2.h[0] = *(const v8bf*)(bp + 2 * 512);
    b2.h[1] = *(const v8bf*)(bp + 2 * 512 + 8);
    b3.h[0] = *(const v8bf*)(bp + 3 * 512);
    b3.h[1] = *(const v8bf*)(bp + 3 * 512 + 8);

    acc0 = __builtin_amdgcn_wmma_f32_16x16x32_bf16(false, a.v, false, b0.v,
                                                   (short)0, acc0, false, false);
    acc1 = __builtin_amdgcn_wmma_f32_16x16x32_bf16(false, a.v, false, b1.v,
                                                   (short)0, acc1, false, false);
    acc2 = __builtin_amdgcn_wmma_f32_16x16x32_bf16(false, a.v, false, b2.v,
                                                   (short)0, acc2, false, false);
    acc3 = __builtin_amdgcn_wmma_f32_16x16x32_bf16(false, a.v, false, b3.v,
                                                   (short)0, acc3, false, false);
  }

  // ---- Epilogue: bias add + store ----
  const float bias0 = conv_b[0 * 16 + m];
  const float bias1 = conv_b[1 * 16 + m];
  const float bias2 = conv_b[2 * 16 + m];
  const float bias3 = conv_b[3 * 16 + m];

  float* outp =
      out + (((size_t)b * OH + (oh0 + wave)) * OW + ow0) * COUT;
#pragma unroll
  for (int r = 0; r < 8; ++r) {
    const int owl = r + 8 * kb;          // D row = output pixel within M-tile
    float* q = outp + (size_t)owl * COUT;
    q[0 * 16 + m] = acc0[r] + bias0;
    q[1 * 16 + m] = acc1[r] + bias1;
    q[2 * 16 + m] = acc2[r] + bias2;
    q[3 * 16 + m] = acc3[r] + bias3;
  }
}

extern "C" void kernel_launch(void* const* d_in, const int* in_sizes, int n_in,
                              void* d_out, int out_size, void* d_ws, size_t ws_size,
                              hipStream_t stream) {
  const float* x      = (const float*)d_in[0];
  const float* conv_w = (const float*)d_in[1];
  const float* conv_b = (const float*)d_in[2];
  float* out = (float*)d_out;
  __bf16* wpk = (__bf16*)d_ws;  // 36,864 B of workspace for packed weights

  pack_weights_kernel<<<dim3(72), 256, 0, stream>>>(conv_w, wpk);

  dim3 grid(OW / TOW, OH / TOH, BATCH);  // 16 x 64 x 16 workgroups
  conv_downsample_kernel<<<grid, 128, 0, stream>>>(x, wpk, conv_b, out);
}
